// HungarianMatcher_44272522887913
// MI455X (gfx1250) — matrix operand order repaired and checked by reading, not compiled, not run on hardware
//
#include <hip/hip_runtime.h>
#include <hip/hip_bf16.h>
#include <stdint.h>

// ---- problem constants (from setup_inputs) ----
#define NQ     600     // bs*Q = 2*300 pred queries
#define NT     60      // targets
#define NC     91      // classes
#define MW     256     // mask width/height
#define HWPX   65536   // 256*256
#define EPS_CLS  1e-8f
#define EPS_DICE 1e-5f

#define NSLICE 16                    // K-slices per row-tile (grid parallelism)
#define NROWT  ((NQ + 15) / 16)      // 38 row tiles
#define SLPX   (HWPX / NSLICE)       // 4096 pixels per slice
#define NSTEP  (SLPX / 2 / 32)       // 64 steps per K-group
#define PARTSZ 3168                  // floats per partial: 3*16*64 dots + 6*16 stats

#define LN2F     0.6931471805599453f
#define INVLN2F  1.4426950408889634f

typedef __attribute__((ext_vector_type(16))) _Float16 v16h;
typedef __attribute__((ext_vector_type(8)))  float    v8f;
typedef __attribute__((ext_vector_type(4)))  _Float16 v4h;

union HU { v16h v; _Float16 h[16]; unsigned short u[16]; uint4 q[2]; };
union CU { v8f  v; float f[8]; };

__device__ __forceinline__ float fast_rcp(float x)  { return __builtin_amdgcn_rcpf(x); }
// raw v_log_f32 (log2); safe here: argument is always >= 1 (never subnormal)
__device__ __forceinline__ float fast_ln_ge1(float x) { return __builtin_amdgcn_logf(x) * LN2F; }

__device__ __forceinline__ float giou_xyxy(float ax, float ay, float az, float aw,
                                           float bx, float by, float bz, float bw) {
    const float area1 = (az - ax) * (aw - ay);
    const float area2 = (bz - bx) * (bw - by);
    const float iw = fminf(az, bz) - fmaxf(ax, bx);
    const float ih = fminf(aw, bw) - fmaxf(ay, by);
    const float inter = fmaxf(iw, 0.f) * fmaxf(ih, 0.f);
    const float uni = area1 + area2 - inter;
    const float iou = inter / uni;
    const float ew = fmaxf(az, bz) - fminf(ax, bx);
    const float eh = fmaxf(aw, bw) - fminf(ay, by);
    const float areae = fmaxf(ew, 0.f) * fmaxf(eh, 0.f);
    return iou - (areae - uni) / areae;
}

// ---------------------------------------------------------------------------
// Kernel 1: focal class cost + L1 bbox cost + GIoU bbox cost -> initialize out
// ---------------------------------------------------------------------------
__global__ __launch_bounds__(64) void hm_base_kernel(
    const float* __restrict__ logits, const float* __restrict__ pboxes,
    const float* __restrict__ tboxes, const int* __restrict__ tids,
    float* __restrict__ out)
{
    const int n = blockIdx.x;
    const int t = threadIdx.x;
    if (t >= NT) return;

    const float lg = logits[n * NC + tids[t]];
    const float pr = fast_rcp(1.f + __expf(-lg));
    const float pos = 0.25f * (1.f - pr) * (1.f - pr) * (-__logf(pr + EPS_CLS));
    const float neg = 0.75f * pr * pr * (-__logf(1.f - pr + EPS_CLS));
    const float cls = pos - neg;

    const float4 pb = *(const float4*)(pboxes + 4 * n);
    const float4 tb = *(const float4*)(tboxes + 4 * t);
    const float l1 = fabsf(pb.x - tb.x) + fabsf(pb.y - tb.y) +
                     fabsf(pb.z - tb.z) + fabsf(pb.w - tb.w);

    const float ax = pb.x - 0.5f * pb.z, ay = pb.y - 0.5f * pb.w;
    const float az = pb.x + 0.5f * pb.z, aw = pb.y + 0.5f * pb.w;
    const float bx = tb.x - 0.5f * tb.z, by = tb.y - 0.5f * tb.w;
    const float bz = tb.x + 0.5f * tb.z, bw = tb.y + 0.5f * tb.w;
    const float g = giou_xyxy(ax, ay, az, aw, bx, by, bz, bw);

    out[n * NT + t] = cls + l1 - g;
}

// ---------------------------------------------------------------------------
// Kernel 2: target-mask sums + masks_to_boxes (+ faithful cxcywh->xyxy quirk)
// ---------------------------------------------------------------------------
__global__ __launch_bounds__(256) void hm_tstats_kernel(
    const int* __restrict__ tmasks, float* __restrict__ tsum, float* __restrict__ tboxc)
{
    const int t = blockIdx.x;
    const int tid = threadIdx.x;
    __shared__ float s[5][256];

    float sm = 0.f, xmx = -1e30f, ymx = -1e30f, xmn = 1e8f, ymn = 1e8f;
    const int* mp = tmasks + (size_t)t * HWPX;
    for (int i = tid; i < HWPX; i += 256) {
        const float m  = (float)mp[i];
        const float cx = (float)(i & (MW - 1));
        const float cy = (float)(i >> 8);
        sm += m;
        xmx = fmaxf(xmx, m * cx); ymx = fmaxf(ymx, m * cy);
        if (m != 0.f) { xmn = fminf(xmn, m * cx); ymn = fminf(ymn, m * cy); }
    }
    s[0][tid] = sm; s[1][tid] = xmx; s[2][tid] = ymx; s[3][tid] = xmn; s[4][tid] = ymn;
    __syncthreads();
    for (int off = 128; off > 0; off >>= 1) {
        if (tid < off) {
            s[0][tid] += s[0][tid + off];
            s[1][tid] = fmaxf(s[1][tid], s[1][tid + off]);
            s[2][tid] = fmaxf(s[2][tid], s[2][tid + off]);
            s[3][tid] = fminf(s[3][tid], s[3][tid + off]);
            s[4][tid] = fminf(s[4][tid], s[4][tid + off]);
        }
        __syncthreads();
    }
    if (tid == 0) {
        tsum[t] = s[0][0];
        const float x0 = s[3][0], y0 = s[4][0], x1 = s[1][0], y1 = s[2][0];
        // reference treats the xyxy result as cxcywh and converts again
        tboxc[4 * t + 0] = x0 - 0.5f * x1;
        tboxc[4 * t + 1] = y0 - 0.5f * y1;
        tboxc[4 * t + 2] = x0 + 0.5f * x1;
        tboxc[4 * t + 3] = y0 + 0.5f * y1;
    }
}

// ---------------------------------------------------------------------------
// Kernel 3: K-sliced fused pass over pred_masks -> partial dots + row stats.
// Grid: (NSLICE, NROWT) = 608 blocks so HBM streaming covers the whole chip.
// Block = 256 threads = 8 waves: 2 K-groups x 4 target column tiles.
// Per pixel: e=exp(-x); d=1+e; p=rcp(d); -log p = ln d; -log(1-p) = x+ln d
//   => 1 v_exp + 1 v_log + 1 v_rcp per pixel, no guards, no divide chain.
// ---------------------------------------------------------------------------
__global__ __launch_bounds__(256) void hm_mask_kernel(
    const float* __restrict__ pmasks, const int* __restrict__ tmasks,
    float* __restrict__ partials)
{
    const int tid  = threadIdx.x;
    const int lane = tid & 31;
    const int wave = tid >> 5;
    const int grp  = wave >> 2;   // K-half within slice
    const int ct   = wave & 3;    // target column tile
    const int sl   = blockIdx.x;  // K-slice
    const int rt   = blockIdx.y;  // row tile
    const int rowbase = rt * 16;

    __shared__ __align__(16) _Float16 sA[2][2][3][16][32]; // [grp][buf][mat][row][k] 12KB
    __shared__ float sAcc[3][16][64];                      // 12KB
    __shared__ float sStat[6][256];                        // 6KB
    __shared__ float sRow[6][16];

    // staging assignment: each thread owns one row, 4 contiguous pixels/step
    const int gt   = tid & 127;
    const int srow = gt >> 3;
    const int scol = (gt & 7) << 2;
    int nrow = rowbase + srow; if (nrow > NQ - 1) nrow = NQ - 1;
    const float* rowptr = pmasks + (size_t)nrow * HWPX;

    float acc_sp = 0.f, acc_sn = 0.f;
    float axmx = -1e30f, aymx = -1e30f, axmn = 1e8f, aymn = 1e8f;

    v8f c0 = {0.f,0.f,0.f,0.f,0.f,0.f,0.f,0.f};
    v8f c1 = c0, c2 = c0;

    const int kgbase = sl * SLPX + grp * (SLPX / 2);
    const int arow   = lane & 15;               // A fragment row (M)
    const int k0     = (lane >> 4) << 4;        // A fragment K base within step
    const int tgt    = ct * 16 + (lane & 15);   // B fragment column (N)
    const int tcl    = (tgt < NT) ? tgt : (NT - 1);
    const unsigned short bscale = (tgt < NT) ? (unsigned short)0x3C00 : (unsigned short)0;
    const int* tgtrow = tmasks + (size_t)tcl * HWPX;

#pragma unroll 2
    for (int step = 0; step < NSTEP; ++step) {
        const int kbase = kgbase + (step << 5);
        const int buf = step & 1;

        // ----- stage: 4 pixels -> p / pos_px / neg_px (f32 math), f16 to LDS -----
        const float4 vv = *(const float4*)(rowptr + kbase + scol);
        v4h hp, hpo, hne;
        {
            const float xv[4] = {vv.x, vv.y, vv.z, vv.w};
            // 32-pixel step never crosses an image row: hoist coords
            const float cy  = (float)(kbase >> 8);
            const float cx0 = (float)((kbase & (MW - 1)) + scol);
            _Float16 a0[4], a1[4], a2[4];
#pragma unroll
            for (int j = 0; j < 4; ++j) {
                const float x  = xv[j];
                const float e  = __builtin_amdgcn_exp2f(x * -INVLN2F); // exp(-x)
                const float d  = 1.f + e;
                const float p  = fast_rcp(d);        // sigmoid(x)
                const float ld = fast_ln_ge1(d);     // = -log(p), d >= 1
                const float om = 1.f - p;
                const float po = 0.25f * om * om * ld;         // pos_px
                const float ne = 0.75f * p * p * (x + ld);     // neg_px (= -log(1-p))
                acc_sp += p;
                acc_sn += ne;
                const float cx = cx0 + (float)j;
                const float mx = x * cx, my = x * cy;
                axmx = fmaxf(axmx, mx); aymx = fmaxf(aymx, my);
                if (x != 0.f) { axmn = fminf(axmn, mx); aymn = fminf(aymn, my); }
                a0[j] = (_Float16)p; a1[j] = (_Float16)po; a2[j] = (_Float16)ne;
            }
            hp.x = a0[0]; hp.y = a0[1]; hp.z = a0[2]; hp.w = a0[3];
            hpo.x = a1[0]; hpo.y = a1[1]; hpo.z = a1[2]; hpo.w = a1[3];
            hne.x = a2[0]; hne.y = a2[1]; hne.z = a2[2]; hne.w = a2[3];
        }
        *(v4h*)(&sA[grp][buf][0][srow][scol]) = hp;
        *(v4h*)(&sA[grp][buf][1][srow][scol]) = hpo;
        *(v4h*)(&sA[grp][buf][2][srow][scol]) = hne;

        __syncthreads();   // double-buffered -> one barrier per step is sufficient

        // ----- B fragment straight from global: 16 contiguous binary ints -----
        HU ub;
        {
            const int* tp = tgtrow + kbase + ((lane >> 4) << 4);
            const int4 q0 = ((const int4*)tp)[0];
            const int4 q1 = ((const int4*)tp)[1];
            const int4 q2 = ((const int4*)tp)[2];
            const int4 q3 = ((const int4*)tp)[3];
            ub.u[0]  = (unsigned short)(q0.x * bscale);
            ub.u[1]  = (unsigned short)(q0.y * bscale);
            ub.u[2]  = (unsigned short)(q0.z * bscale);
            ub.u[3]  = (unsigned short)(q0.w * bscale);
            ub.u[4]  = (unsigned short)(q1.x * bscale);
            ub.u[5]  = (unsigned short)(q1.y * bscale);
            ub.u[6]  = (unsigned short)(q1.z * bscale);
            ub.u[7]  = (unsigned short)(q1.w * bscale);
            ub.u[8]  = (unsigned short)(q2.x * bscale);
            ub.u[9]  = (unsigned short)(q2.y * bscale);
            ub.u[10] = (unsigned short)(q2.z * bscale);
            ub.u[11] = (unsigned short)(q2.w * bscale);
            ub.u[12] = (unsigned short)(q3.x * bscale);
            ub.u[13] = (unsigned short)(q3.y * bscale);
            ub.u[14] = (unsigned short)(q3.z * bscale);
            ub.u[15] = (unsigned short)(q3.w * bscale);
        }

        // ----- A fragments: two ds_load_b128 per matrix (contiguous k = 16h + s, matching B) -----
        HU ua0, ua1, ua2;
        ua0.q[0] = *(const uint4*)(&sA[grp][buf][0][arow][k0]);
        ua0.q[1] = *(const uint4*)(&sA[grp][buf][0][arow][k0 + 8]);
        ua1.q[0] = *(const uint4*)(&sA[grp][buf][1][arow][k0]);
        ua1.q[1] = *(const uint4*)(&sA[grp][buf][1][arow][k0 + 8]);
        ua2.q[0] = *(const uint4*)(&sA[grp][buf][2][arow][k0]);
        ua2.q[1] = *(const uint4*)(&sA[grp][buf][2][arow][k0 + 8]);

        c0 = __builtin_amdgcn_wmma_f32_16x16x32_f16(false, ua0.v, false, ub.v, (short)0, c0, false, false);
        c1 = __builtin_amdgcn_wmma_f32_16x16x32_f16(false, ua1.v, false, ub.v, (short)0, c1, false, false);
        c2 = __builtin_amdgcn_wmma_f32_16x16x32_f16(false, ua2.v, false, ub.v, (short)0, c2, false, false);
    }

    // ----- reduce across the block's two K-groups -----
    sStat[0][tid] = acc_sp; sStat[1][tid] = acc_sn;
    sStat[2][tid] = axmx;   sStat[3][tid] = aymx;
    sStat[4][tid] = axmn;   sStat[5][tid] = aymn;

    const int ncol  = ct * 16 + (lane & 15);
    const int mbase = (lane >> 4) << 3;
    CU u0, u1, u2; u0.v = c0; u1.v = c1; u2.v = c2;
    if (grp == 0) {
#pragma unroll
        for (int r = 0; r < 8; ++r) {
            sAcc[0][mbase + r][ncol] = u0.f[r];
            sAcc[1][mbase + r][ncol] = u1.f[r];
            sAcc[2][mbase + r][ncol] = u2.f[r];
        }
    }
    __syncthreads();
    if (grp == 1) {
#pragma unroll
        for (int r = 0; r < 8; ++r) {
            sAcc[0][mbase + r][ncol] += u0.f[r];
            sAcc[1][mbase + r][ncol] += u1.f[r];
            sAcc[2][mbase + r][ncol] += u2.f[r];
        }
    }
    if (tid < 16) {  // 16 contributors per row: 8 lanes x 2 groups
        float sp = 0.f, sn = 0.f;
        float xmx = -1e30f, ymx = -1e30f, xmn = 1e8f, ymn = 1e8f;
#pragma unroll
        for (int g = 0; g < 2; ++g)
#pragma unroll
            for (int j = 0; j < 8; ++j) {
                const int id = g * 128 + tid * 8 + j;
                sp += sStat[0][id]; sn += sStat[1][id];
                xmx = fmaxf(xmx, sStat[2][id]); ymx = fmaxf(ymx, sStat[3][id]);
                xmn = fminf(xmn, sStat[4][id]); ymn = fminf(ymn, sStat[5][id]);
            }
        sRow[0][tid] = sp; sRow[1][tid] = sn;
        sRow[2][tid] = xmx; sRow[3][tid] = ymx;
        sRow[4][tid] = xmn; sRow[5][tid] = ymn;
    }
    __syncthreads();

    // ----- write this block's partial (deterministic; no atomics) -----
    float* pb = partials + ((size_t)rt * NSLICE + sl) * PARTSZ;
    const float* accf = (const float*)sAcc;
    for (int i = tid; i < 3072; i += 256) pb[i] = accf[i];
    if (tid < 96) pb[3072 + tid] = ((const float*)sRow)[tid];
}

// ---------------------------------------------------------------------------
// Kernel 4: combine K-slice partials + fused dice/focal/mask-GIoU epilogue
// ---------------------------------------------------------------------------
__global__ __launch_bounds__(256) void hm_combine_kernel(
    const float* __restrict__ partials, const float* __restrict__ tsum,
    const float* __restrict__ tboxc, float* __restrict__ out)
{
    const int rt  = blockIdx.x;
    const int tid = threadIdx.x;
    const int rowbase = rt * 16;

    __shared__ float cAcc[3][16][64];
    __shared__ float cRow[6][16];

    const float* base = partials + (size_t)rt * NSLICE * PARTSZ;

    float* accf = (float*)cAcc;
    for (int i = tid; i < 3072; i += 256) {
        float s = 0.f;
#pragma unroll
        for (int sl = 0; sl < NSLICE; ++sl) s += base[(size_t)sl * PARTSZ + i];
        accf[i] = s;
    }
    if (tid < 96) {
        const int st = tid >> 4;       // stat id
        float v = base[3072 + tid];
#pragma unroll
        for (int sl = 1; sl < NSLICE; ++sl) {
            const float w = base[(size_t)sl * PARTSZ + 3072 + tid];
            if (st < 2)      v += w;
            else if (st < 4) v = fmaxf(v, w);
            else             v = fminf(v, w);
        }
        ((float*)cRow)[tid] = v;
    }
    __syncthreads();

    for (int pi = tid; pi < 16 * NT; pi += 256) {
        const int row = pi / NT;
        const int t   = pi - row * NT;
        const int n   = rowbase + row;
        if (n < NQ) {
            const float inter = cAcc[0][row][t];
            const float posd  = cAcc[1][row][t];
            const float negd  = cAcc[2][row][t];
            const float sp = cRow[0][row], sn = cRow[1][row];
            const float dice  = 1.f - (2.f * inter + EPS_DICE) / (sp + tsum[t] + EPS_DICE);
            const float focal = (posd + sn - negd) * (1.f / (float)HWPX);

            const float bxmn = cRow[4][row], bymn = cRow[5][row];
            const float bxmx = cRow[2][row], bymx = cRow[3][row];
            const float ax = bxmn - 0.5f * bxmx, ay = bymn - 0.5f * bymx;
            const float az = bxmn + 0.5f * bxmx, aw = bymn + 0.5f * bymx;
            const float4 tb = *(const float4*)(tboxc + 4 * t);
            const float g = giou_xyxy(ax, ay, az, aw, tb.x, tb.y, tb.z, tb.w);

            out[n * NT + t] += dice + focal - g;
        }
    }
}

// ---------------------------------------------------------------------------
extern "C" void kernel_launch(void* const* d_in, const int* in_sizes, int n_in,
                              void* d_out, int out_size, void* d_ws, size_t ws_size,
                              hipStream_t stream) {
    (void)in_sizes; (void)n_in; (void)out_size; (void)ws_size;
    const float* logits = (const float*)d_in[0];   // [2,300,91]
    const float* pboxes = (const float*)d_in[1];   // [2,300,4]
    const float* pmasks = (const float*)d_in[2];   // [2,300,256,256]
    const float* tboxes = (const float*)d_in[3];   // [60,4]
    const int*   tids   = (const int*)d_in[4];     // [60]
    const int*   tmasks = (const int*)d_in[5];     // [60,256,256]
    float* out = (float*)d_out;                    // [2,300,60]

    float* tsum     = (float*)d_ws;                // 64 floats
    float* tboxc    = tsum + 64;                   // 240 floats (converted boxes)
    float* partials = tsum + 512;                  // 38*16*PARTSZ floats (~7.7 MB)

    hipLaunchKernelGGL(hm_base_kernel,    dim3(NQ), dim3(64),  0, stream,
                       logits, pboxes, tboxes, tids, out);
    hipLaunchKernelGGL(hm_tstats_kernel,  dim3(NT), dim3(256), 0, stream,
                       tmasks, tsum, tboxc);
    hipLaunchKernelGGL(hm_mask_kernel,    dim3(NSLICE, NROWT), dim3(256), 0, stream,
                       pmasks, tmasks, partials);
    hipLaunchKernelGGL(hm_combine_kernel, dim3(NROWT), dim3(256), 0, stream,
                       partials, tsum, tboxc, out);
}
